// Attention_36498632081881
// MI455X (gfx1250) — compile-verified, matrix operand order checked by reading, not run on hardware
//
#include <hip/hip_runtime.h>

// ---------------------------------------------------------------- constants
#define H_    8
#define D_    64
#define DM_   512
#define NSEQ  2048
#define BATCH 2
#define EPS_  1e-5f
#define QKVO  (3 * H_ * D_)   // 1536

typedef __bf16 bf16_t;
typedef __attribute__((ext_vector_type(16))) __bf16 v16bf;
typedef __attribute__((ext_vector_type(8)))  __bf16 v8bf;
typedef __attribute__((ext_vector_type(8)))  float  v8f;
typedef __attribute__((ext_vector_type(4)))  int    v4i;

// Pointer flavors for the async LDS-DMA builtin:
// diag shows param0 = "int __vector(4) __device__ *"  (addrspace(1))
typedef __attribute__((address_space(1))) v4i* gv4i_p;   // global int4*
typedef __attribute__((address_space(3))) v4i* sv4i_p;   // LDS int4*

#if defined(__has_builtin)
#  if __has_builtin(__builtin_amdgcn_global_load_async_to_lds_b128)
#    define HAVE_ASYNC_COPY 1
#  else
#    define HAVE_ASYNC_COPY 0
#  endif
#else
#  define HAVE_ASYNC_COPY 0
#endif

// ---------------------------------------------------------------- helpers
__device__ __forceinline__ bf16_t f2bf(float f) {
    unsigned u = __builtin_bit_cast(unsigned, f);
    unsigned r = (u + 0x7FFFu + ((u >> 16) & 1u)) >> 16;   // round-nearest-even
    unsigned short s = (unsigned short)r;
    return __builtin_bit_cast(bf16_t, s);
}

// Stage `bytes` (multiple of 2048) of contiguous global memory into LDS with the
// whole 128-thread workgroup. Uses CDNA5 async LDS-DMA (ASYNCcnt) when the
// toolchain exposes it; falls back to regular LDS stores otherwise. Ends with a
// workgroup barrier so staged data is visible to all waves.
__device__ __forceinline__ void stage_strip(bf16_t* __restrict__ lds_dst,
                                            const bf16_t* __restrict__ gsrc,
                                            int bytes) {
    for (int off = threadIdx.x * 16; off < bytes; off += 128 * 16) {
#if HAVE_ASYNC_COPY
        __builtin_amdgcn_global_load_async_to_lds_b128(
            (gv4i_p)((const char*)gsrc + off),
            (sv4i_p)((char*)lds_dst + off),
            0, 0);
#else
        *(v8bf*)((char*)lds_dst + off) = *(const v8bf*)((const char*)gsrc + off);
#endif
    }
#if HAVE_ASYNC_COPY
#  if __has_builtin(__builtin_amdgcn_s_wait_asynccnt)
    __builtin_amdgcn_s_wait_asynccnt(0);
#  else
    asm volatile("s_wait_asynccnt 0x0" ::: "memory");
#  endif
#endif
    __syncthreads();
}

// Load one 16(row) x 32(col) bf16 fragment from a row-major matrix (global or
// LDS; LDS pointers lower to ds_load_b128). ISA 16-bit A layout: lanes 0-15 ->
// row = lane, elems 0..7 = K 0..7, elems 8..15 = K 16..23; lanes 16-31: K +8.
__device__ __forceinline__ v16bf load_tile16x32(const bf16_t* __restrict__ base, int ld) {
    const int lane = threadIdx.x & 31;
    const int r    = lane & 15;
    const int kh   = (lane >> 4) << 3;          // 0 or 8
    const bf16_t* p = base + (size_t)r * ld + kh;
    v8bf lo = *(const v8bf*)(p);                // K = kh .. kh+7
    v8bf hi = *(const v8bf*)(p + 16);           // K = 16+kh .. 16+kh+7
    v16bf out;
#pragma unroll
    for (int i = 0; i < 8; ++i) { out[i] = lo[i]; out[i + 8] = hi[i]; }
    return out;
}

__device__ __forceinline__ v8f wmma_bf16(v16bf a, v16bf b, v8f c) {
    return __builtin_amdgcn_wmma_f32_16x16x32_bf16(
        /*neg_a=*/false, a, /*neg_b=*/false, b,
        /*c_mod=*/(short)0, c, /*reuse_a=*/false, /*reuse_b=*/false);
}

// ---------------------------------------------------------------- converters
__global__ void k_cvt_bf16(const float* __restrict__ src, bf16_t* __restrict__ dst, size_t n) {
    size_t i = (size_t)blockIdx.x * 256 + threadIdx.x;
    if (i < n) dst[i] = f2bf(src[i]);
}

// src [K x O] row-major f32  ->  dst [O x K] bf16 (i.e. transposed)
__global__ void k_transpose_cvt(const float* __restrict__ src, bf16_t* __restrict__ dst,
                                int K, int O) {
    size_t i = (size_t)blockIdx.x * 256 + threadIdx.x;
    if (i < (size_t)K * O) {
        int o  = (int)(i % O);
        int kk = (int)(i / O);
        dst[(size_t)o * K + kk] = f2bf(src[i]);
    }
}

// ---------------------------------------------------------------- QKV GEMM
// C[i,o] = sum_k xbf[i,k] * w_qkv[k,o] ; B^T = wqkvT[o,k].
// 4 waves, block = 16 rows x 64 cols. The shared 16x512 A strip is staged into
// LDS (async DMA) once and reused by all 4 waves; B streams from global.
__global__ __launch_bounds__(128) void k_qkv_gemm(
    const bf16_t* __restrict__ xbf, const bf16_t* __restrict__ wqkvT,
    bf16_t* __restrict__ qbf, bf16_t* __restrict__ kbf, bf16_t* __restrict__ vT) {
    __shared__ __align__(16) bf16_t As[16 * DM_];          // 16 KB

    const int rows16 = (BATCH * NSEQ) / 16;
    int itile = blockIdx.x % rows16;
    int ot    = blockIdx.x / rows16;
    int wave  = threadIdx.x >> 5;
    int lane  = threadIdx.x & 31;
    int o0    = ot * 64 + wave * 16;

    stage_strip(As, xbf + (size_t)itile * 16 * DM_, 16 * DM_ * (int)sizeof(bf16_t));

    const bf16_t* brow = wqkvT + (size_t)o0 * DM_;

    v8f acc = {};
    for (int kk = 0; kk < DM_; kk += 32) {
        if (kk + 32 < DM_) __builtin_prefetch(brow + kk + 32, 0, 3);
        v16bf a = load_tile16x32(As + kk, DM_);            // ds_load_b128
        v16bf b = load_tile16x32(brow + kk, DM_);          // global_load_b128
        acc = wmma_bf16(a, b, acc);
    }

    int r0 = (lane >> 4) << 3;
    int nc = lane & 15;
    int ocol = o0 + nc;
    int t   = ocol / DM_;          // 0=q 1=k 2=v (split axis)
    int rem = ocol % DM_;
    int h   = rem / D_;
    int d   = rem % D_;
#pragma unroll
    for (int r = 0; r < 8; ++r) {
        int grow = itile * 16 + r0 + r;
        int b_ = grow / NSEQ, i = grow % NSEQ;
        size_t hb = (size_t)b_ * H_ + h;
        bf16_t v = f2bf(acc[r]);
        if (t == 0)      qbf[(hb * NSEQ + i) * D_ + d] = v;
        else if (t == 1) kbf[(hb * NSEQ + i) * D_ + d] = v;
        else             vT [(hb * D_   + d) * NSEQ + i] = v;
    }
}

// ---------------------------------------------------------------- scores+softmax
// One workgroup (4 waves) per (b,h, 16-row block): S = scale * Q K^T via WMMA
// into a 16x2048 f32 LDS strip, then row softmax, write P (f32) to d_out attn.
__global__ __launch_bounds__(128) void k_scores_softmax(
    const bf16_t* __restrict__ qbf, const bf16_t* __restrict__ kbf,
    float* __restrict__ attn) {
    __shared__ float S[16][NSEQ];                          // 128 KB (WGP: 320 KB)
    __shared__ float sinv[16];
    __shared__ __align__(16) bf16_t Qs[16 * D_];           // 2 KB

    int itile = blockIdx.x % (NSEQ / 16);
    int bh    = blockIdx.x / (NSEQ / 16);
    const bf16_t* q    = qbf + ((size_t)bh * NSEQ + itile * 16) * D_;
    const bf16_t* kmat = kbf + (size_t)bh * NSEQ * D_;
    float* outp = attn + (size_t)bh * NSEQ * NSEQ + (size_t)itile * 16 * NSEQ;

    int wave = threadIdx.x >> 5;
    int lane = threadIdx.x & 31;
    int r0 = (lane >> 4) << 3;
    int nc = lane & 15;

    stage_strip(Qs, q, 16 * D_ * (int)sizeof(bf16_t));

    v16bf a0 = load_tile16x32(Qs, D_);
    v16bf a1 = load_tile16x32(Qs + 32, D_);
    const float scale = 0.125f;          // D^-0.5

    const int jbeg = wave * (NSEQ / 4), jend = jbeg + NSEQ / 4;
    for (int j = jbeg; j < jend; j += 16) {
        if (j + 16 < jend) __builtin_prefetch(kmat + (size_t)(j + 16) * D_, 0, 3);
        v16bf b0 = load_tile16x32(kmat + (size_t)j * D_, D_);
        v16bf b1 = load_tile16x32(kmat + (size_t)j * D_ + 32, D_);
        v8f c = {};
        c = wmma_bf16(a0, b0, c);
        c = wmma_bf16(a1, b1, c);
#pragma unroll
        for (int r = 0; r < 8; ++r) S[r0 + r][j + nc] = c[r] * scale;
    }
    __syncthreads();

    int row = threadIdx.x >> 3;          // 16 rows, 8 threads each
    int tt  = threadIdx.x & 7;
    float mx = -3.0e38f;
    for (int j = tt; j < NSEQ; j += 8) mx = fmaxf(mx, S[row][j]);
#pragma unroll
    for (int o = 1; o < 8; o <<= 1) mx = fmaxf(mx, __shfl_xor(mx, o));
    float sum = 0.f;
    for (int j = tt; j < NSEQ; j += 8) {
        float e = __expf(S[row][j] - mx);
        S[row][j] = e;
        sum += e;
    }
#pragma unroll
    for (int o = 1; o < 8; o <<= 1) sum += __shfl_xor(sum, o);
    if (tt == 0) sinv[row] = 1.0f / sum;
    __syncthreads();

    for (int u = threadIdx.x; u < 16 * NSEQ; u += 128) {
        int r  = u >> 11;                // NSEQ == 2048
        int c2 = u & (NSEQ - 1);
        outp[u] = S[r][c2] * sinv[r];
    }
}

// ---------------------------------------------------------------- heads mix + LN
// attn[g] = LN_h( sum_h P[h] * head_mix[h][g] ) * g + b  (in-place on d_out);
// also emits a bf16 copy for the PV GEMM.
__global__ __launch_bounds__(256) void k_mix_ln(
    float* __restrict__ attn, const float* __restrict__ head_mix,
    const float* __restrict__ lng, const float* __restrict__ lnb,
    bf16_t* __restrict__ attn_bf) {
    __shared__ float hm[H_ * H_];
    __shared__ float gg[H_], bb[H_];
    if (threadIdx.x < H_ * H_) hm[threadIdx.x] = head_mix[threadIdx.x];
    if (threadIdx.x < H_) { gg[threadIdx.x] = lng[threadIdx.x]; bb[threadIdx.x] = lnb[threadIdx.x]; }
    __syncthreads();

    size_t idx  = (size_t)blockIdx.x * 256 + threadIdx.x;
    size_t NN   = (size_t)NSEQ * NSEQ;
    size_t bidx = idx / NN;
    size_t ij   = idx % NN;

    float p[H_], m[H_];
#pragma unroll
    for (int h = 0; h < H_; ++h) p[h] = attn[(bidx * H_ + h) * NN + ij];
#pragma unroll
    for (int g = 0; g < H_; ++g) {
        float s = 0.f;
#pragma unroll
        for (int h = 0; h < H_; ++h) s += p[h] * hm[h * H_ + g];
        m[g] = s;
    }
    float mean = 0.f;
#pragma unroll
    for (int g = 0; g < H_; ++g) mean += m[g];
    mean *= (1.0f / H_);
    float var = 0.f;
#pragma unroll
    for (int g = 0; g < H_; ++g) { float d = m[g] - mean; var += d * d; }
    float inv = rsqrtf(var * (1.0f / H_) + EPS_);
#pragma unroll
    for (int g = 0; g < H_; ++g) {
        float v = (m[g] - mean) * inv * gg[g] + bb[g];
        size_t off = (bidx * H_ + g) * NN + ij;
        attn[off]    = v;
        attn_bf[off] = f2bf(v);
    }
}

// ---------------------------------------------------------------- P @ V
// o[b,i,h*64+d] = sum_j P[b,h,i,j] V[b,h,j,d];  B^T = vT[b,h,D,n] row-major.
// The 16x2048 attn strip (64 KB) is staged into LDS once and reused by all 4
// waves (each wave owns a different 16-wide d slice of the output).
__global__ __launch_bounds__(128) void k_attn_v(
    const bf16_t* __restrict__ attn_bf, const bf16_t* __restrict__ vT,
    bf16_t* __restrict__ obf) {
    __shared__ __align__(16) bf16_t As[16 * NSEQ];         // 64 KB

    int itile = blockIdx.x % (NSEQ / 16);
    int bh    = blockIdx.x / (NSEQ / 16);
    int b_ = bh >> 3, h = bh & 7;
    int wave = threadIdx.x >> 5;
    int lane = threadIdx.x & 31;
    int d0   = wave * 16;

    stage_strip(As, attn_bf + ((size_t)bh * NSEQ + itile * 16) * NSEQ,
                16 * NSEQ * (int)sizeof(bf16_t));

    const bf16_t* brow = vT + ((size_t)bh * D_ + d0) * NSEQ;

    v8f acc = {};
    for (int j = 0; j < NSEQ; j += 32) {
        if (j + 32 < NSEQ) __builtin_prefetch(brow + j + 32, 0, 3);
        v16bf a = load_tile16x32(As + j, NSEQ);            // ds_load_b128
        v16bf b = load_tile16x32(brow + j, NSEQ);
        acc = wmma_bf16(a, b, acc);
    }

    int r0 = (lane >> 4) << 3;
    int nc = lane & 15;
#pragma unroll
    for (int r = 0; r < 8; ++r) {
        int i = itile * 16 + r0 + r;
        obf[((size_t)b_ * NSEQ + i) * DM_ + h * D_ + d0 + nc] = f2bf(acc[r]);
    }
}

// ---------------------------------------------------------------- out-proj + LN + residual sum
// C = o @ w_out + b_out ; LN over features ; + x ; partial row-sum per block.
__global__ __launch_bounds__(128) void k_out_proj(
    const bf16_t* __restrict__ obf, const bf16_t* __restrict__ woutT,
    const float* __restrict__ b_out, const float* __restrict__ lng,
    const float* __restrict__ lnb, const float* __restrict__ x,
    float* __restrict__ partial) {
    __shared__ float sm[16][DM_];                          // 32 KB
    __shared__ __align__(16) bf16_t As[16 * DM_];          // 16 KB

    int itile = blockIdx.x % (NSEQ / 16);
    int b_    = blockIdx.x / (NSEQ / 16);
    size_t row0 = (size_t)b_ * NSEQ + (size_t)itile * 16;
    int wave = threadIdx.x >> 5;
    int lane = threadIdx.x & 31;

    stage_strip(As, obf + row0 * DM_, 16 * DM_ * (int)sizeof(bf16_t));

    v8f acc[8] = {};
    for (int kk = 0; kk < DM_; kk += 32) {
        v16bf a = load_tile16x32(As + kk, DM_);            // ds_load_b128
#pragma unroll
        for (int t = 0; t < 8; ++t) {
            v16bf b = load_tile16x32(woutT + (size_t)(wave * 128 + t * 16) * DM_ + kk, DM_);
            acc[t] = wmma_bf16(a, b, acc[t]);
        }
    }

    int r0 = (lane >> 4) << 3;
    int nc = lane & 15;
#pragma unroll
    for (int t = 0; t < 8; ++t) {
        int col = wave * 128 + t * 16 + nc;
        float bo = b_out[col];
#pragma unroll
        for (int r = 0; r < 8; ++r) sm[r0 + r][col] = acc[t][r] + bo;
    }
    __syncthreads();

    int row = threadIdx.x >> 3;
    int tt  = threadIdx.x & 7;
    float s = 0.f;
    for (int j = tt; j < DM_; j += 8) s += sm[row][j];
#pragma unroll
    for (int o = 1; o < 8; o <<= 1) s += __shfl_xor(s, o);
    float mean = s * (1.0f / DM_);
    float vs = 0.f;
    for (int j = tt; j < DM_; j += 8) { float d = sm[row][j] - mean; vs += d * d; }
#pragma unroll
    for (int o = 1; o < 8; o <<= 1) vs += __shfl_xor(vs, o);
    float inv = rsqrtf(vs * (1.0f / DM_) + EPS_);

    const float* xrow = x + (row0 + row) * DM_;
    for (int j = tt; j < DM_; j += 8)
        sm[row][j] = (sm[row][j] - mean) * inv * lng[j] + lnb[j] + xrow[j];
    __syncthreads();

    for (int col = threadIdx.x; col < DM_; col += 128) {
        float cs = 0.f;
#pragma unroll
        for (int r = 0; r < 16; ++r) cs += sm[r][col];
        partial[(size_t)blockIdx.x * DM_ + col] = cs;
    }
}

// Deterministic final reduction over the 128 row-blocks of each batch.
__global__ void k_reduce(const float* __restrict__ partial, float* __restrict__ out) {
    int i = blockIdx.x * 256 + threadIdx.x;
    if (i < BATCH * DM_) {
        int b_ = i / DM_, o = i % DM_;
        float s = 0.f;
        for (int it = 0; it < NSEQ / 16; ++it)
            s += partial[((size_t)b_ * (NSEQ / 16) + it) * DM_ + o];
        out[i] = s;
    }
}

// ---------------------------------------------------------------- launch
extern "C" void kernel_launch(void* const* d_in, const int* in_sizes, int n_in,
                              void* d_out, int out_size, void* d_ws, size_t ws_size,
                              hipStream_t stream) {
    const float* x        = (const float*)d_in[0];
    const float* w_qkv    = (const float*)d_in[1];
    const float* head_mix = (const float*)d_in[2];
    const float* ln_h_g   = (const float*)d_in[3];
    const float* ln_h_b   = (const float*)d_in[4];
    const float* w_out    = (const float*)d_in[5];
    const float* b_out    = (const float*)d_in[6];
    const float* ln_o_g   = (const float*)d_in[7];
    const float* ln_o_b   = (const float*)d_in[8];

    float* out  = (float*)d_out;                       // [2, 512]
    float* attn = out + (size_t)BATCH * DM_;           // [2, 8, 2048, 2048]

    // ---- workspace carve-up (~158 MB) ----
    char* w = (char*)d_ws;
    bf16_t* xbf     = (bf16_t*)w; w += (size_t)BATCH * NSEQ * DM_ * sizeof(bf16_t);
    bf16_t* wqkvT   = (bf16_t*)w; w += (size_t)DM_ * QKVO * sizeof(bf16_t);
    bf16_t* woutT   = (bf16_t*)w; w += (size_t)DM_ * DM_ * sizeof(bf16_t);
    bf16_t* qbf     = (bf16_t*)w; w += (size_t)BATCH * H_ * NSEQ * D_ * sizeof(bf16_t);
    bf16_t* kbf     = (bf16_t*)w; w += (size_t)BATCH * H_ * NSEQ * D_ * sizeof(bf16_t);
    bf16_t* vT      = (bf16_t*)w; w += (size_t)BATCH * H_ * D_ * NSEQ * sizeof(bf16_t);
    bf16_t* attn_bf = (bf16_t*)w; w += (size_t)BATCH * H_ * NSEQ * NSEQ * sizeof(bf16_t);
    bf16_t* obf     = (bf16_t*)w; w += (size_t)BATCH * NSEQ * DM_ * sizeof(bf16_t);
    float*  partial = (float*)w;  w += (size_t)BATCH * (NSEQ / 16) * DM_ * sizeof(float);

    // ---- precision conversion / weight transposes ----
    {
        size_t n = (size_t)BATCH * NSEQ * DM_;
        k_cvt_bf16<<<(unsigned)((n + 255) / 256), 256, 0, stream>>>(x, xbf, n);
    }
    k_transpose_cvt<<<(DM_ * QKVO + 255) / 256, 256, 0, stream>>>(w_qkv, wqkvT, DM_, QKVO);
    k_transpose_cvt<<<(DM_ * DM_ + 255) / 256, 256, 0, stream>>>(w_out, woutT, DM_, DM_);

    // ---- pipeline ----
    k_qkv_gemm<<<((BATCH * NSEQ) / 16) * (QKVO / 64), 128, 0, stream>>>(
        xbf, wqkvT, qbf, kbf, vT);

    k_scores_softmax<<<BATCH * H_ * (NSEQ / 16), 128, 0, stream>>>(qbf, kbf, attn);

    k_mix_ln<<<(unsigned)(((size_t)BATCH * NSEQ * NSEQ) / 256), 256, 0, stream>>>(
        attn, head_mix, ln_h_g, ln_h_b, attn_bf);

    k_attn_v<<<BATCH * H_ * (NSEQ / 16), 128, 0, stream>>>(attn_bf, vT, obf);

    k_out_proj<<<BATCH * (NSEQ / 16), 128, 0, stream>>>(
        obf, woutT, b_out, ln_o_g, ln_o_b, x, partial);

    k_reduce<<<(BATCH * DM_ + 255) / 256, 256, 0, stream>>>(partial, out);
}
